// RefineNet_42923903156750
// MI455X (gfx1250) — compile-verified, matrix operand order checked by reading, not run on hardware
//
#include <hip/hip_runtime.h>
#include <hip/hip_bf16.h>

typedef __attribute__((ext_vector_type(16))) _Float16 v16h;
typedef __attribute__((ext_vector_type(8)))  float    v8f;

#define BN_EPS 1e-5f
#define BATCHN 4
#define MAXKCH 14              // max K-chunks (dec: 16*27=432 -> 14)

// ---------------------------------------------------------------------------
// Repack f32 OIDHW weights -> f16 fragment-native order [chunk][lane][16].
// (K pattern: lanes 0-15 hold K={0..7,16..23}; lanes 16-31 hold K={8..15,24..31})
// K-padding and N-padding are ZERO -> padded A entries only need safe addresses.
// ---------------------------------------------------------------------------
__global__ void prep_weights(const float* __restrict__ w, _Float16* __restrict__ dst,
                             int ci, int co, int kpad) {
  int idx = blockIdx.x * blockDim.x + threadIdx.x;
  if (idx >= kpad * 16) return;
  int kk = idx >> 4;
  int n  = idx & 15;
  float v = 0.f;
  int K = ci * 27;
  if (kk < K && n < co) {
    int c = kk / 27, t = kk - c * 27;
    v = w[((size_t)n * ci + c) * 27 + t];
  }
  int chunk = kk >> 5;
  int k32   = kk & 31;
  int half  = (k32 >> 3) & 1;
  int grp   = k32 >> 4;
  int j     = grp * 8 + (k32 & 7);
  int lane  = half * 16 + n;
  dst[(size_t)chunk * 512 + (size_t)lane * 16 + j] = (_Float16)v;
}

// ---------------------------------------------------------------------------
// Per-K metadata + precomputed element delta.
// meta: [1:0]=kd [3:2]=kh [5:4]=kw [6]=sel
// delta: cc*sites + (kd-1)*HWd + (kh-1)*Wd + (kw-1)*Od
//   conv : (HWd,Wd,Od) = (H*W, W, 1)   (full spatial delta precomputed)
//   convT: (HWd,Wd,Od) = (0,0,0)       (spatial computed at runtime, parity)
// K-padding entries: center tap, delta 0 -> always-safe address, zero weight.
// ---------------------------------------------------------------------------
__global__ void prep_ktab(int2* __restrict__ tab, int c0, int c1, int kpad,
                          int sites, int HWd, int Wd, int Od) {
  int kk = blockIdx.x * blockDim.x + threadIdx.x;
  if (kk >= kpad) return;
  int ktot = (c0 + c1) * 27;
  int2 e = make_int2(1 | (1 << 2) | (1 << 4), 0);   // safe center tap
  if (kk < ktot) {
    int c = kk / 27, t = kk - c * 27;
    int kd = t / 9, r = t - kd * 9;
    int kh = r / 3, kw = r - kh * 3;
    int sel = (c >= c0) ? 1 : 0;
    int cc  = sel ? (c - c0) : c;
    e.x = kd | (kh << 2) | (kw << 4) | (sel << 6);
    e.y = cc * sites + (kd - 1) * HWd + (kh - 1) * Wd + (kw - 1) * Od;
  }
  tab[kk] = e;
}

// ---------------------------------------------------------------------------
// Forward conv3d 3x3x3 SAME as implicit GEMM on WMMA (f16 in, f16 out).
// Row tiles: one wave = 16 consecutive w in one (d,h) row x 16 cout.
// grid = (ceil(H*segs/4), D, B); segs = W/16 (po2, shift = ssh).
// dB = (src1 - src0) in elements, folded into deltas at LDS-fill time.
// Interior tiles (all taps in-bounds, uniform test) take a compare-free path.
// ---------------------------------------------------------------------------
__global__ __launch_bounds__(128) void conv3d_wmma(
    const _Float16* __restrict__ src, int c0, int dB, int cout,
    const _Float16* __restrict__ wf16, const int2* __restrict__ ktab,
    _Float16* __restrict__ dst,
    int D_, int H_, int W_, int ssh, int kchunks)
{
  __shared__ _Float16 s_w[MAXKCH * 512];
  __shared__ int      s_meta[512];
  __shared__ int      s_delta[512];

  const int tid = threadIdx.x;
  {
    const int* wsrc = (const int*)wf16;
    int* wdst = (int*)s_w;
    for (int i = tid; i < kchunks * 256; i += 128) wdst[i] = wsrc[i];
    for (int i = tid; i < kchunks * 32;  i += 128) {
      int2 e = ktab[i];
      s_meta[i]  = e.x;
      s_delta[i] = e.y + (((e.x >> 6) & 1) ? dB : 0);
    }
  }
  __syncthreads();

  const int segs = 1 << ssh;
  const int t = blockIdx.x * 4 + (tid >> 5);          // wave-uniform row tile
  if (t >= H_ * segs) return;
  const int h  = t >> ssh;
  const int wb = (t & (segs - 1)) << 4;
  const int d  = blockIdx.y;
  const int b  = blockIdx.z;

  const int HW    = H_ * W_;
  const int sites = D_ * HW;
  const _Float16* base = src + (long)b * c0 * sites;  // uniform base

  const int lane = tid & 31;
  const int half = lane >> 4;
  const int m    = lane & 15;
  const int ww0  = wb + m;
  const int siteOff = d * HW + h * W_ + ww0;          // per-lane 32-bit

  const bool interior = (d >= 1) & (d + 1 < D_) & (h >= 1) & (h + 1 < H_) &
                        (wb >= 16) & (wb + 32 <= W_);

  v8f acc = {};
  if (interior) {
    // -------- compare-free interior path --------
    for (int kc = 0; kc < kchunks; ++kc) {
      int dl[16];
      const int kb0 = (kc << 5) + (half << 3);
      *(int4*)&dl[0]  = *(const int4*)&s_delta[kb0];
      *(int4*)&dl[4]  = *(const int4*)&s_delta[kb0 + 4];
      *(int4*)&dl[8]  = *(const int4*)&s_delta[kb0 + 16];
      *(int4*)&dl[12] = *(const int4*)&s_delta[kb0 + 20];
      v16h a;
#pragma unroll
      for (int j = 0; j < 16; ++j)
        a[j] = base[(unsigned)(siteOff + dl[j])];
      v16h bf = *(const v16h*)(s_w + (size_t)kc * 512 + (size_t)lane * 16);
      acc = __builtin_amdgcn_wmma_f32_16x16x32_f16(false, a, false, bf,
                                                   (short)0, acc, false, false);
    }
  } else {
    // -------- checked border path (branch-free per element) --------
    for (int kc = 0; kc < kchunks; ++kc) {
      int mk[16], dl[16];
      const int kb0 = (kc << 5) + (half << 3);
      *(int4*)&mk[0]  = *(const int4*)&s_meta[kb0];
      *(int4*)&mk[4]  = *(const int4*)&s_meta[kb0 + 4];
      *(int4*)&mk[8]  = *(const int4*)&s_meta[kb0 + 16];
      *(int4*)&mk[12] = *(const int4*)&s_meta[kb0 + 20];
      *(int4*)&dl[0]  = *(const int4*)&s_delta[kb0];
      *(int4*)&dl[4]  = *(const int4*)&s_delta[kb0 + 4];
      *(int4*)&dl[8]  = *(const int4*)&s_delta[kb0 + 16];
      *(int4*)&dl[12] = *(const int4*)&s_delta[kb0 + 20];
      v16h a;
#pragma unroll
      for (int j = 0; j < 16; ++j) {
        int e  = mk[j];
        int dd = d   + (e & 3)        - 1;
        int hh = h   + ((e >> 2) & 3) - 1;
        int ww = ww0 + ((e >> 4) & 3) - 1;
        unsigned ok = (unsigned)((unsigned)dd < (unsigned)D_) &
                      (unsigned)((unsigned)hh < (unsigned)H_) &
                      (unsigned)((unsigned)ww < (unsigned)W_);
        unsigned msk  = 0u - ok;                       // keep load unconditional
        unsigned offs = ((unsigned)(siteOff + dl[j])) & msk;
        _Float16 v = base[offs];
        a[j] = ok ? v : (_Float16)0;
      }
      v16h bf = *(const v16h*)(s_w + (size_t)kc * 512 + (size_t)lane * 16);
      acc = __builtin_amdgcn_wmma_f32_16x16x32_f16(false, a, false, bf,
                                                   (short)0, acc, false, false);
    }
  }

  const int n  = lane & 15;
  const int mb = half << 3;
  if (n < cout) {
    long off = ((long)b * cout + n) * sites + (long)d * HW + (long)h * W_ + wb + mb;
    union { _Float16 h8[8]; float4 v4; } u;
#pragma unroll
    for (int r = 0; r < 8; ++r) u.h8[r] = (_Float16)acc[r];
    *(float4*)(dst + off) = u.v4;
  }
}

// ---------------------------------------------------------------------------
// Transposed conv3d k=3, s=2 (out = 2x in) as gather GEMM (f16 in/out).
// Tap valid iff q = o + k - 1 even and 0<=q<2*in-1; input coord = q/2.
// Parity masking is inherent (per-lane w parity) -> single branch-free path.
// ---------------------------------------------------------------------------
__global__ __launch_bounds__(128) void convT3d_wmma(
    const _Float16* __restrict__ src, int c0, int dB, int cout,
    const _Float16* __restrict__ wf16, const int2* __restrict__ ktab,
    _Float16* __restrict__ dst,
    int Di, int Hi, int Wi, int ssh, int kchunks)
{
  __shared__ _Float16 s_w[MAXKCH * 512];
  __shared__ int      s_meta[512];
  __shared__ int      s_delta[512];

  const int tid = threadIdx.x;
  {
    const int* wsrc = (const int*)wf16;
    int* wdst = (int*)s_w;
    for (int i = tid; i < kchunks * 256; i += 128) wdst[i] = wsrc[i];
    for (int i = tid; i < kchunks * 32;  i += 128) {
      int2 e = ktab[i];
      s_meta[i]  = e.x;
      s_delta[i] = e.y + (((e.x >> 6) & 1) ? dB : 0);
    }
  }
  __syncthreads();

  const int Do = Di << 1, Ho = Hi << 1, Wo = Wi << 1;
  const int segs = 1 << ssh;
  const int t = blockIdx.x * 4 + (tid >> 5);
  if (t >= Ho * segs) return;
  const int h  = t >> ssh;
  const int wb = (t & (segs - 1)) << 4;
  const int d  = blockIdx.y;
  const int b  = blockIdx.z;

  const int HWo    = Ho * Wo;
  const int sitesO = Do * HWo;
  const int HWi    = Hi * Wi;
  const int sitesI = Di * HWi;
  const unsigned limD = (unsigned)(2 * Di - 1);
  const unsigned limH = (unsigned)(2 * Hi - 1);
  const unsigned limW = (unsigned)(2 * Wi - 1);
  const _Float16* base = src + (long)b * c0 * sitesI;

  const int lane = tid & 31;
  const int half = lane >> 4;
  const int m    = lane & 15;
  const int ww0  = wb + m;

  v8f acc = {};
  for (int kc = 0; kc < kchunks; ++kc) {
    int mk[16], dl[16];
    const int kb0 = (kc << 5) + (half << 3);
    *(int4*)&mk[0]  = *(const int4*)&s_meta[kb0];
    *(int4*)&mk[4]  = *(const int4*)&s_meta[kb0 + 4];
    *(int4*)&mk[8]  = *(const int4*)&s_meta[kb0 + 16];
    *(int4*)&mk[12] = *(const int4*)&s_meta[kb0 + 20];
    *(int4*)&dl[0]  = *(const int4*)&s_delta[kb0];
    *(int4*)&dl[4]  = *(const int4*)&s_delta[kb0 + 4];
    *(int4*)&dl[8]  = *(const int4*)&s_delta[kb0 + 16];
    *(int4*)&dl[12] = *(const int4*)&s_delta[kb0 + 20];

    v16h a;
#pragma unroll
    for (int j = 0; j < 16; ++j) {
      int e  = mk[j];
      int qd = d   + (e & 3)        - 1;
      int qh = h   + ((e >> 2) & 3) - 1;
      int qw = ww0 + ((e >> 4) & 3) - 1;
      unsigned ok = (unsigned)((~qd) & 1) & (unsigned)((unsigned)qd < limD) &
                    (unsigned)((~qh) & 1) & (unsigned)((unsigned)qh < limH) &
                    (unsigned)((~qw) & 1) & (unsigned)((unsigned)qw < limW);
      int sp = (qd >> 1) * HWi + (qh >> 1) * Wi + (qw >> 1);
      unsigned msk  = 0u - ok;                        // keep load unconditional
      unsigned offs = ((unsigned)(dl[j] + sp)) & msk;
      _Float16 v = base[offs];
      a[j] = ok ? v : (_Float16)0;
    }
    v16h bf = *(const v16h*)(s_w + (size_t)kc * 512 + (size_t)lane * 16);
    acc = __builtin_amdgcn_wmma_f32_16x16x32_f16(false, a, false, bf,
                                                 (short)0, acc, false, false);
  }

  const int n  = lane & 15;
  const int mb = half << 3;
  if (n < cout) {
    long off = ((long)b * cout + n) * sitesO + (long)d * HWo + (long)h * Wo + wb + mb;
    union { _Float16 h8[8]; float4 v4; } u;
#pragma unroll
    for (int r = 0; r < 8; ++r) u.h8[r] = (_Float16)acc[r];
    *(float4*)(dst + off) = u.v4;
  }
}

// ---------------------------------------------------------------------------
// Elementwise helpers (f16 activations)
// ---------------------------------------------------------------------------
__global__ void f32_to_f16(const float* __restrict__ s, _Float16* __restrict__ d, long n) {
  long i = (long)blockIdx.x * blockDim.x + threadIdx.x;
  if (i < n) d[i] = (_Float16)s[i];
}

__global__ void zero_f32(float* p, int n) {
  int i = blockIdx.x * blockDim.x + threadIdx.x;
  if (i < n) p[i] = 0.f;
}

__global__ void bn_stats(const _Float16* __restrict__ x, float* __restrict__ stats,
                         int C, long S) {
  int bc = blockIdx.x;                 // b*C + c
  int c  = bc % C;
  const _Float16* p = x + (size_t)bc * S;
  float s = 0.f, q = 0.f;
  for (long i = (long)blockIdx.y * blockDim.x + threadIdx.x; i < S;
       i += (long)gridDim.y * blockDim.x) {
    float v = (float)p[i];
    s += v; q += v * v;
  }
  __shared__ float sh[256], shq[256];
  sh[threadIdx.x] = s; shq[threadIdx.x] = q;
  __syncthreads();
  for (int o = 128; o > 0; o >>= 1) {
    if (threadIdx.x < o) {
      sh[threadIdx.x]  += sh[threadIdx.x + o];
      shq[threadIdx.x] += shq[threadIdx.x + o];
    }
    __syncthreads();
  }
  if (threadIdx.x == 0) {
    atomicAdd(&stats[2 * c],     sh[0]);
    atomicAdd(&stats[2 * c + 1], shq[0]);
  }
}

// grid = (ceil(S/256), C, B)
__global__ void bn_apply_relu(_Float16* __restrict__ x, const float* __restrict__ stats,
                              const float* __restrict__ g, const float* __restrict__ bta,
                              int Bn, int C, long S) {
  long i = (long)blockIdx.x * blockDim.x + threadIdx.x;
  if (i >= S) return;
  int c = blockIdx.y;
  int b = blockIdx.z;
  float cnt = (float)Bn * (float)S;
  float mu  = stats[2 * c] / cnt;
  float var = stats[2 * c + 1] / cnt - mu * mu;
  _Float16* p = x + ((long)b * C + c) * S;
  float y = ((float)p[i] - mu) * rsqrtf(var + BN_EPS) * g[c] + bta[c];
  p[i] = (_Float16)fmaxf(y, 0.f);
}

// grid = (ceil(Ho*Wo/256), Do, BC); Wo po2 (shift = wsh)
__global__ void maxpool2(const _Float16* __restrict__ in, _Float16* __restrict__ out,
                         int Di, int Hi, int Wi, int wsh) {
  int Ho = Hi >> 1, Wo = Wi >> 1;
  int i = blockIdx.x * blockDim.x + threadIdx.x;
  if (i >= Ho * Wo) return;
  int wo = i & (Wo - 1);
  int ho = i >> wsh;
  int dd = blockIdx.y;
  int bc = blockIdx.z;
  const _Float16* p = in + (size_t)bc * Di * Hi * Wi;
  float mx = -__builtin_inff();
#pragma unroll
  for (int dz = 0; dz < 2; ++dz)
#pragma unroll
    for (int hy = 0; hy < 2; ++hy)
#pragma unroll
      for (int wx = 0; wx < 2; ++wx)
        mx = fmaxf(mx, (float)p[(long)(2 * dd + dz) * Hi * Wi + (2 * ho + hy) * Wi + (2 * wo + wx)]);
  out[(size_t)bc * (Di >> 1) * Ho * Wo + (size_t)dd * Ho * Wo + i] = (_Float16)mx;
}

// ---------------------------------------------------------------------------
// dispconv (4->1 ch, 3x3x3 SAME, bias) + tanh on first 192*640 flat elements
// ---------------------------------------------------------------------------
__global__ void dispconv_tanh(const _Float16* __restrict__ hbuf, const float* __restrict__ w,
                              const float* __restrict__ bias, float* __restrict__ out,
                              int Bn) {
  const int IMGHW = 192 * 640;
  const int D_ = 32, H_ = 160, W_ = 256;
  const int HW = H_ * W_;
  long idx = (long)blockIdx.x * blockDim.x + threadIdx.x;
  if (idx >= (long)Bn * IMGHW) return;
  int b  = (int)(idx / IMGHW);
  int r  = (int)(idx % IMGHW);
  int d  = r / HW;
  int r2 = r % HW;
  int hh = r2 >> 8;          // / 256
  int ww = r2 & 255;         // % 256
  float sum = bias[0];
#pragma unroll
  for (int c = 0; c < 4; ++c) {
    const _Float16* plane = hbuf + ((long)b * 4 + c) * D_ * HW;
    const float* wc = w + c * 27;
#pragma unroll
    for (int kd = 0; kd < 3; ++kd) {
      int dd = d + kd - 1;
      if ((unsigned)dd >= (unsigned)D_) continue;
#pragma unroll
      for (int kh = 0; kh < 3; ++kh) {
        int h2 = hh + kh - 1;
        if ((unsigned)h2 >= (unsigned)H_) continue;
#pragma unroll
        for (int kw = 0; kw < 3; ++kw) {
          int w2 = ww + kw - 1;
          if ((unsigned)w2 >= (unsigned)W_) continue;
          sum += (float)plane[(long)dd * HW + h2 * W_ + w2] * wc[(kd * 3 + kh) * 3 + kw];
        }
      }
    }
  }
  out[idx] = tanhf(sum);
}

// ---------------------------------------------------------------------------
// Host orchestration
// ---------------------------------------------------------------------------
static inline int ilog2i(int x) { int s = 0; while ((1 << s) < x) ++s; return s; }

extern "C" void kernel_launch(void* const* d_in, const int* in_sizes, int n_in,
                              void* d_out, int out_size, void* d_ws, size_t ws_size,
                              hipStream_t stream) {
  (void)in_sizes; (void)n_in; (void)out_size; (void)ws_size;

  const float* X = (const float*)d_in[0];
  const float *encW[5], *encG[5], *encB[5];
  for (int i = 0; i < 5; ++i) {
    encW[i] = (const float*)d_in[1 + 3 * i];
    encG[i] = (const float*)d_in[2 + 3 * i];
    encB[i] = (const float*)d_in[3 + 3 * i];
  }
  const float *decW[5], *decG[5], *decB[5];      // by stage 0..4 (input order 4..0)
  for (int k = 0; k < 5; ++k) {
    int st = 4 - k;
    decW[st] = (const float*)d_in[16 + 3 * k];
    decG[st] = (const float*)d_in[17 + 3 * k];
    decB[st] = (const float*)d_in[18 + 3 * k];
  }
  const float* dispW = (const float*)d_in[31];
  const float* dispB = (const float*)d_in[32];

  // ---- workspace carve-up ----
  size_t off = 0;
  char* wsb = (char*)d_ws;
  auto carve = [&](size_t bytes) -> char* {
    char* p = wsb + off;
    off += (bytes + 255) & ~(size_t)255;
    return p;
  };
  const long S0 = 32L * 160 * 256;
  _Float16* xh   = (_Float16*)carve((size_t)BATCHN * 4 * S0 * sizeof(_Float16));
  _Float16* big  = (_Float16*)carve((size_t)BATCHN * 4 * S0 * sizeof(_Float16));
  _Float16* pong = (_Float16*)carve((size_t)BATCHN * 4 * (S0 / 8) * sizeof(_Float16));
  const int fd[5][3] = {{16,80,128},{8,40,64},{4,20,32},{2,10,16},{1,5,8}};
  const int fc[5]    = {4, 4, 6, 8, 16};
  _Float16* feat[5];
  for (int i = 0; i < 5; ++i) {
    long fS = (long)fd[i][0] * fd[i][1] * fd[i][2];
    feat[i] = (_Float16*)carve((size_t)BATCHN * fc[i] * fS * sizeof(_Float16));
  }
  float* stats = (float*)carve(32 * sizeof(float));

  const int encCi[5] = {4, 4, 4, 6, 8},    encCo[5] = {4, 4, 6, 8, 16};
  const int decCi[5] = {8, 8, 12, 16, 16}, decCo[5] = {4, 4, 4, 6, 8};
  auto kpad_of = [](int ci) { return ((ci * 27 + 31) / 32) * 32; };
  _Float16 *wqE[5], *wqD[5];
  int2 *ktE[5], *ktD[5];
  for (int i = 0; i < 5; ++i) {
    wqE[i] = (_Float16*)carve((size_t)kpad_of(encCi[i]) * 16 * sizeof(_Float16));
    ktE[i] = (int2*)carve((size_t)kpad_of(encCi[i]) * sizeof(int2));
  }
  for (int i = 0; i < 5; ++i) {
    wqD[i] = (_Float16*)carve((size_t)kpad_of(decCi[i]) * 16 * sizeof(_Float16));
    ktD[i] = (int2*)carve((size_t)kpad_of(decCi[i]) * sizeof(int2));
  }

  // ---- input f32 -> f16 ----
  long nX = (long)BATCHN * 4 * S0;
  f32_to_f16<<<(unsigned)((nX + 255) / 256), 256, 0, stream>>>(X, xh, nX);

  // ---- repack weights + K-tables ----
  const int ed[5][3] = {{32,160,256},{16,80,128},{8,40,64},{4,20,32},{2,10,16}};
  for (int i = 0; i < 5; ++i) {
    int kp = kpad_of(encCi[i]);
    prep_weights<<<(kp * 16 + 255) / 256, 256, 0, stream>>>(encW[i], wqE[i], encCi[i], encCo[i], kp);
    int D = ed[i][0], H = ed[i][1], W = ed[i][2];
    prep_ktab<<<(kp + 255) / 256, 256, 0, stream>>>(ktE[i], encCi[i], 0, kp,
                                                    D * H * W, H * W, W, 1);
  }
  const int decC0[5] = {4, 4, 6, 8, 16};   // decoder src0 channels
  const int decC1[5] = {4, 4, 6, 8, 0};    // decoder skip channels
  const int dd_in[5][3] = {{16,80,128},{8,40,64},{4,20,32},{2,10,16},{1,5,8}};
  for (int i = 0; i < 5; ++i) {
    int kp = kpad_of(decCi[i]);
    prep_weights<<<(kp * 16 + 255) / 256, 256, 0, stream>>>(decW[i], wqD[i], decCi[i], decCo[i], kp);
    int sI = dd_in[i][0] * dd_in[i][1] * dd_in[i][2];
    prep_ktab<<<(kp + 255) / 256, 256, 0, stream>>>(ktD[i], decC0[i], decC1[i], kp,
                                                    sI, 0, 0, 0);
  }

  auto run_bn = [&](_Float16* buf, const float* g, const float* bta, int C, long S) {
    zero_f32<<<1, 64, 0, stream>>>(stats, 32);
    dim3 gs((unsigned)(BATCHN * C), 64);
    bn_stats<<<gs, 256, 0, stream>>>(buf, stats, C, S);
    dim3 ga((unsigned)((S + 255) / 256), (unsigned)C, BATCHN);
    bn_apply_relu<<<ga, 256, 0, stream>>>(buf, stats, g, bta, BATCHN, C, S);
  };

  // ---- encoder ----
  const _Float16* cur = xh;
  for (int i = 0; i < 5; ++i) {
    int D = ed[i][0], H = ed[i][1], W = ed[i][2];
    long sites = (long)D * H * W;
    int kch = (encCi[i] * 27 + 31) / 32;
    int ssh = ilog2i(W >> 4);
    dim3 gc((unsigned)((H * (W >> 4) + 3) / 4), (unsigned)D, BATCHN);
    conv3d_wmma<<<gc, 128, 0, stream>>>(cur, encCi[i], 0, encCo[i],
                                        wqE[i], ktE[i], big, D, H, W, ssh, kch);
    run_bn(big, encG[i], encB[i], encCo[i], sites);
    int Ho = H >> 1, Wo = W >> 1;
    dim3 gp((unsigned)((Ho * Wo + 255) / 256), (unsigned)(D >> 1), (unsigned)(BATCHN * encCo[i]));
    maxpool2<<<gp, 256, 0, stream>>>(big, feat[i], D, H, W, ilog2i(Wo));
    cur = feat[i];
  }

  auto run_convT = [&](const _Float16* s0, const _Float16* s1, int c0, int co,
                       const _Float16* wq, const int2* kt, _Float16* dst,
                       int Di, int Hi, int Wi, int kch) {
    int dB = s1 ? (int)(s1 - s0) : 0;
    int Ho = Hi << 1, Wo = Wi << 1;
    int ssh = ilog2i(Wo >> 4);
    dim3 g((unsigned)((Ho * (Wo >> 4) + 3) / 4), (unsigned)(Di << 1), BATCHN);
    convT3d_wmma<<<g, 128, 0, stream>>>(s0, c0, dB, co, wq, kt, dst,
                                        Di, Hi, Wi, ssh, kch);
  };

  // ---- decoder (ping-pong big <-> pong) ----
  run_convT(feat[4], nullptr, 16, 8, wqD[4], ktD[4], big, 1, 5, 8, 14);   // h4 (2,10,16)x8
  run_bn(big, decG[4], decB[4], 8, 2L * 10 * 16);
  run_convT(big, feat[3], 8, 6, wqD[3], ktD[3], pong, 2, 10, 16, 14);     // h3 (4,20,32)x6
  run_bn(pong, decG[3], decB[3], 6, 4L * 20 * 32);
  run_convT(pong, feat[2], 6, 4, wqD[2], ktD[2], big, 4, 20, 32, 11);     // h2 (8,40,64)x4
  run_bn(big, decG[2], decB[2], 4, 8L * 40 * 64);
  run_convT(big, feat[1], 4, 4, wqD[1], ktD[1], pong, 8, 40, 64, 7);      // h1 (16,80,128)x4
  run_bn(pong, decG[1], decB[1], 4, 16L * 80 * 128);
  run_convT(pong, feat[0], 4, 4, wqD[0], ktD[0], big, 16, 80, 128, 7);    // h0 (32,160,256)x4
  run_bn(big, decG[0], decB[0], 4, S0);

  // ---- disp head + tanh -> d_out (B,1,192,640) ----
  long nout = (long)BATCHN * 192 * 640;
  dispconv_tanh<<<(unsigned)((nout + 255) / 256), 256, 0, stream>>>(
      big, dispW, dispB, (float*)d_out, BATCHN);
}
